// ActorNetwork_75995151336066
// MI455X (gfx1250) — matrix operand order repaired and checked by reading, not compiled, not run on hardware
//
#include <hip/hip_runtime.h>
#include <hip/hip_bf16.h>

typedef __attribute__((ext_vector_type(16))) __bf16 v16bf;
typedef __attribute__((ext_vector_type(8)))  float  v8f;
typedef unsigned short u16;
typedef unsigned int   u32;

// GIDX[i*7+k] = flat index into A row (64 wide) for gate of node i, neighbor slot k
__constant__ int cGIDX[56] = {
  1, 2, 3,32,40,48,56,
  1,17,25,33,41,49,57,
  2,10,26,34,42,50,58,
  3,11,19,35,43,51,59,
  4,12,20,28,44,52,60,
  5,13,21,29,37,53,61,
  6,14,22,30,38,46,62,
  7,15,23,31,39,47,55
};

__device__ __forceinline__ u16 f2bf(float f) {
  u32 u = __float_as_uint(f);
  u32 r = u + 0x7fffu + ((u >> 16) & 1u);   // round-to-nearest-even
  return (u16)(r >> 16);
}
__device__ __forceinline__ float bf2f(u16 h) {
  return __uint_as_float(((u32)h) << 16);
}

// unpack 8 bf16 (uint4) -> 8 f32
__device__ __forceinline__ void unpack8(uint4 q, float* f) {
  u32 w[4] = {q.x, q.y, q.z, q.w};
#pragma unroll
  for (int t = 0; t < 4; ++t) {
    f[2 * t + 0] = __uint_as_float(w[t] << 16);
    f[2 * t + 1] = __uint_as_float(w[t] & 0xffff0000u);
  }
}
// pack 8 f32 -> 8 bf16 (uint4)
__device__ __forceinline__ uint4 pack8(const float* f) {
  u32 w[4];
#pragma unroll
  for (int t = 0; t < 4; ++t) {
    u32 lo = f2bf(f[2 * t + 0]);
    u32 hi = f2bf(f[2 * t + 1]);
    w[t] = lo | (hi << 16);
  }
  uint4 q; q.x = w[0]; q.y = w[1]; q.z = w[2]; q.w = w[3];
  return q;
}

union Frag16 {
  v16bf v;
  uint4 q[2];
};

__device__ __forceinline__ v8f wmma_bf16(const Frag16& a, const Frag16& b, v8f c) {
  // D = A(16x32) * B(32x16) + C, bf16 in / f32 acc
  return __builtin_amdgcn_wmma_f32_16x16x32_bf16(false, a.v, false, b.v, (short)0, c,
                                                 false, false);
}

// ---------------------------------------------------------------------------
// Prepack f32 weights (8, Ksrc, Nsrc) row-major into WMMA B-fragment order:
// dst[((i*KT + kt)*NT + nt)*512 + lane*16 + e] = W[i][k][n] (bf16), where
//   n = nt*16 + (lane&15),  k = kt*32 + (e<8 ? (lane>>4)*8 + e : 16 + (lane>>4)*8 + e-8)
// Padded-K space maps through chunks: kp -> chunk = kp/CHP, d = kp%CHP,
//   ksrc = chunk*CH + d   (valid iff d < CH)   [pads per-neighbor blocks]
// ---------------------------------------------------------------------------
__global__ void prepack_kernel(const float* __restrict__ src, u16* __restrict__ dst,
                               int Ksrc, int Nsrc, int KT, int NT, int CH, int CHP,
                               int total)
{
  int idx = blockIdx.x * blockDim.x + threadIdx.x;
  if (idx >= total) return;
  int e  = idx & 15;
  int l  = (idx >> 4) & 31;
  int fr = idx >> 9;
  int nt = fr % NT;
  int kt = (fr / NT) % KT;
  int i  = fr / (NT * KT);
  int kp = kt * 32 + (e < 8 ? (l >> 4) * 8 + e : 16 + (l >> 4) * 8 + (e - 8));
  int n  = nt * 16 + (l & 15);
  int chunk = kp / CHP;
  int d  = kp % CHP;
  int ks = chunk * CH + d;
  float val = 0.f;
  if (d < CH && ks < Ksrc && n < Nsrc)
    val = src[((size_t)i * Ksrc + ks) * Nsrc + n];
  dst[idx] = f2bf(val);
}

// ---------------------------------------------------------------------------
// One fused layer (256 threads = 8 wave32):
// gating GEMM -> softmax*gates -> neighbor aggregation -> main GEMM ->
// LayerNorm -> ReLU -> Znext (bf16, LDS)
// Zc layout:  [r][j][0..DPIN)  bf16     Xt layout: [i][r][0..DPIN) bf16
// hb layout:  [i][r][0..DOUT)  f32 (reused as [i][r][16] logits in phase G)
// Zn layout:  [r][i][0..DOUT)  bf16
// ---------------------------------------------------------------------------
template<int DPIN, int DOUT>
__device__ void layer_step(u16* Zc, u16* Xt, float* hb, u16* Zn,
    const u16* __restrict__ Wp, const u16* __restrict__ Wgp,
    const float* __restrict__ bias, const float* __restrict__ gain,
    const float* __restrict__ beta, const float* __restrict__ bg,
    const float* __restrict__ A, int b0, int tid)
{
  const int lane = tid & 31;
  const int wave = tid >> 5;
  const int KT_G = (7 * DPIN) / 32;

  // ---- Phase G: gating GEMM, logits -> hb[i][r][0..15] (one node per wave) ----
  {
    const int r   = lane & 15;
    const int hk  = lane >> 4;
    const int col = lane & 15;
    const int i   = wave;
    v8f acc = {0.f,0.f,0.f,0.f,0.f,0.f,0.f,0.f};
    for (int kt = 0; kt < KT_G; ++kt) {
      int c    = (kt * 32) / DPIN;           // neighbor slot
      int j    = c + (c >= i ? 1 : 0);       // actual node index
      int doff = (kt * 32) % DPIN;
      Frag16 a, b;
      const u16* za = Zc + (r * 8 + j) * DPIN + doff + hk * 8;
      a.q[0] = *(const uint4*)(za);
      a.q[1] = *(const uint4*)(za + 16);
      const uint4* wb = (const uint4*)(Wgp + (size_t)(i * KT_G + kt) * 512 + lane * 16);
      b.q[0] = wb[0]; b.q[1] = wb[1];
      acc = wmma_bf16(a, b, acc);
    }
    float bgv = (col < 7) ? bg[i * 7 + col] : 0.f;
#pragma unroll
    for (int v = 0; v < 8; ++v)
      hb[(i * 16 + v + hk * 8) * 16 + col] = acc[v] + bgv;
  }
  __syncthreads();

  // ---- Phase S: softmax*gates + aggregation, X = Z + NZ ----
  // pair = tid>>1 covers (i,r); the two lanes of a pair split the d-range.
  {
    const int pair = tid >> 1;
    const int half = tid & 1;
    const int i = pair >> 4;
    const int r = pair & 15;
    float lg[7];
    float m = -1e30f;
#pragma unroll
    for (int k = 0; k < 7; ++k) { lg[k] = hb[(i * 16 + r) * 16 + k]; m = fmaxf(m, lg[k]); }
    float s = 0.f;
#pragma unroll
    for (int k = 0; k < 7; ++k) { lg[k] = expf(lg[k] - m); s += lg[k]; }
    float inv = 1.f / s;
    float cw[7]; int nb[7];
#pragma unroll
    for (int k = 0; k < 7; ++k) {
      nb[k] = k + (k >= i ? 1 : 0);
      float gate = A[(size_t)(b0 + r) * 64 + cGIDX[i * 7 + k]];
      cw[k] = lg[k] * inv * gate;
    }
    const u16* zr = Zc + r * 8 * DPIN;
    u16* xr = Xt + (i * 16 + r) * DPIN;
    const int d0beg = half * (DPIN / 2);
    const int d0end = d0beg + (DPIN / 2);
    for (int d0 = d0beg; d0 < d0end; d0 += 8) {
      float x[8];
      unpack8(*(const uint4*)(zr + i * DPIN + d0), x);
#pragma unroll
      for (int k = 0; k < 7; ++k) {
        float t[8];
        unpack8(*(const uint4*)(zr + nb[k] * DPIN + d0), t);
#pragma unroll
        for (int e = 0; e < 8; ++e) x[e] = fmaf(cw[k], t[e], x[e]);
      }
      *(uint4*)(xr + d0) = pack8(x);
    }
  }
  __syncthreads();

  // ---- Phase M: main GEMM h = X @ W_i + bias ----
  {
    const int KT  = DPIN / 32;
    const int NT  = DOUT / 16;
    const int r   = lane & 15;
    const int hk  = lane >> 4;
    const int col = lane & 15;
    for (int t = wave; t < 8 * NT; t += 8) {
      int i = t / NT;
      int n = t % NT;
      v8f acc = {0.f,0.f,0.f,0.f,0.f,0.f,0.f,0.f};
      for (int kt = 0; kt < KT; ++kt) {
        Frag16 a, b;
        const u16* xa = Xt + (i * 16 + r) * DPIN + kt * 32 + hk * 8;
        a.q[0] = *(const uint4*)(xa);
        a.q[1] = *(const uint4*)(xa + 16);
        const uint4* wb = (const uint4*)(Wp + (size_t)((i * KT + kt) * NT + n) * 512 + lane * 16);
        b.q[0] = wb[0]; b.q[1] = wb[1];
        acc = wmma_bf16(a, b, acc);
      }
      float bv = bias[i * DOUT + n * 16 + col];
#pragma unroll
      for (int v = 0; v < 8; ++v)
        hb[(i * 16 + v + hk * 8) * DOUT + n * 16 + col] = acc[v] + bv;
    }
  }
  __syncthreads();

  // ---- Phase L: LayerNorm + ReLU -> Znext (pair-split, shfl_xor combine) ----
  {
    const int pair = tid >> 1;
    const int half = tid & 1;
    const int i = pair >> 4;
    const int r = pair & 15;
    float* hr = hb + (i * 16 + r) * DOUT;
    const int o0 = half * (DOUT / 2);
    const int o1 = o0 + (DOUT / 2);
    float s = 0.f, sq = 0.f;
    for (int o = o0; o < o1; o += 4) {
      float4 h4 = *(const float4*)(hr + o);
      s  += h4.x + h4.y + h4.z + h4.w;
      sq += h4.x * h4.x + h4.y * h4.y + h4.z * h4.z + h4.w * h4.w;
    }
    s  += __shfl_xor(s, 1);
    sq += __shfl_xor(sq, 1);
    float mu  = s * (1.f / DOUT);
    float var = sq * (1.f / DOUT) - mu * mu;
    float rinv = rsqrtf(var + 1e-5f);
    for (int o = o0; o < o1; o += 8) {
      float zv[8];
      float4 ha = *(const float4*)(hr + o);
      float4 hbv = *(const float4*)(hr + o + 4);
      float4 ga = *(const float4*)(gain + i * DOUT + o);
      float4 gb = *(const float4*)(gain + i * DOUT + o + 4);
      float4 ba = *(const float4*)(beta + i * DOUT + o);
      float4 bb = *(const float4*)(beta + i * DOUT + o + 4);
      zv[0] = fmaxf((ha.x - mu) * rinv * ga.x + ba.x, 0.f);
      zv[1] = fmaxf((ha.y - mu) * rinv * ga.y + ba.y, 0.f);
      zv[2] = fmaxf((ha.z - mu) * rinv * ga.z + ba.z, 0.f);
      zv[3] = fmaxf((ha.w - mu) * rinv * ga.w + ba.w, 0.f);
      zv[4] = fmaxf((hbv.x - mu) * rinv * gb.x + bb.x, 0.f);
      zv[5] = fmaxf((hbv.y - mu) * rinv * gb.y + bb.y, 0.f);
      zv[6] = fmaxf((hbv.z - mu) * rinv * gb.z + bb.z, 0.f);
      zv[7] = fmaxf((hbv.w - mu) * rinv * gb.w + bb.w, 0.f);
      *(uint4*)(Zn + (r * 8 + i) * DOUT + o) = pack8(zv);
    }
  }
  __syncthreads();
}

// ---- Output head: out = tanh(Z(64) @ Wout_i + bout), DOUT padded 5->16 ----
__device__ void out_step(u16* Zc, float* hb, const u16* __restrict__ Wp,
                         const float* __restrict__ bo, float* __restrict__ out,
                         int b0, int tid)
{
  const int lane = tid & 31;
  const int wave = tid >> 5;
  {
    const int r    = lane & 15;
    const int hk   = lane >> 4;
    const int col  = lane & 15;
    const int i    = wave;
    v8f acc = {0.f,0.f,0.f,0.f,0.f,0.f,0.f,0.f};
    for (int kt = 0; kt < 2; ++kt) {
      Frag16 a, b;
      const u16* za = Zc + (r * 8 + i) * 64 + kt * 32 + hk * 8;
      a.q[0] = *(const uint4*)(za);
      a.q[1] = *(const uint4*)(za + 16);
      const uint4* wb = (const uint4*)(Wp + (size_t)(i * 2 + kt) * 512 + lane * 16);
      b.q[0] = wb[0]; b.q[1] = wb[1];
      acc = wmma_bf16(a, b, acc);
    }
    float bv = (col < 5) ? bo[i * 5 + col] : 0.f;
#pragma unroll
    for (int v = 0; v < 8; ++v)
      hb[(i * 16 + v + hk * 8) * 16 + col] = acc[v] + bv;
  }
  __syncthreads();
  if (tid < 128) {
    const int i = tid >> 4;
    const int r = tid & 15;
#pragma unroll
    for (int o = 0; o < 5; ++o)
      out[((size_t)(b0 + r) * 8 + i) * 5 + o] = tanhf(hb[(i * 16 + r) * 16 + o]);
  }
}

// ---------------------------------------------------------------------------
// Fused network kernel: 1 block (8 waves) = 16 batch rows through all layers.
// LDS carve (bytes), 240KB total:
//   L0: Zc@0(8K)      X@8K(8K)     h@16K(128K)  Zn@144K(64K)
//   L1: Zc@144K(64K)  X@0(64K)     h@64K(64K)   Zn@208K(32K)
//   L2: Zc@208K(32K)  X@0(32K)     h@32K(32K)   Zn@64K(16K)
//   Out:Zc@64K(16K)   h@0(8K)
// ---------------------------------------------------------------------------
__global__ __launch_bounds__(256) void actor_fused(
    const float* __restrict__ Z1, const float* __restrict__ Z2,
    const float* __restrict__ Z3, const float* __restrict__ Z4,
    const float* __restrict__ Z5, const float* __restrict__ Z6,
    const float* __restrict__ Z7, const float* __restrict__ Z8,
    const float* __restrict__ A,
    const u16* __restrict__ Wp0, const u16* __restrict__ Wgp0,
    const float* __restrict__ b0_, const float* __restrict__ g0_,
    const float* __restrict__ be0_, const float* __restrict__ bg0_,
    const u16* __restrict__ Wp1, const u16* __restrict__ Wgp1,
    const float* __restrict__ b1_, const float* __restrict__ g1_,
    const float* __restrict__ be1_, const float* __restrict__ bg1_,
    const u16* __restrict__ Wp2, const u16* __restrict__ Wgp2,
    const float* __restrict__ b2_, const float* __restrict__ g2_,
    const float* __restrict__ be2_, const float* __restrict__ bg2_,
    const u16* __restrict__ WpO, const float* __restrict__ bo_,
    float* __restrict__ out)
{
  extern __shared__ char smem[];
  const int tid = threadIdx.x;
  const int b0  = blockIdx.x * 16;

  u16*  Zc0 = (u16*)(smem + 0);
  u16*  X0  = (u16*)(smem + 8192);
  float* H0 = (float*)(smem + 16384);
  u16*  Zs1 = (u16*)(smem + 147456);
  u16*  X1  = (u16*)(smem + 0);
  float* H1 = (float*)(smem + 65536);
  u16*  Zs2 = (u16*)(smem + 212992);
  u16*  X2  = (u16*)(smem + 0);
  float* H2 = (float*)(smem + 32768);
  u16*  Zs3 = (u16*)(smem + 65536);
  float* H3 = (float*)(smem + 0);

  // Build layer-0 Z tile: [r][j][0..32) bf16, d<3 real, rest zero
  if (tid < 128) {
    const float* Zsrc[8] = {Z1, Z2, Z3, Z4, Z5, Z6, Z7, Z8};
    const int r = tid >> 3;
    const int j = tid & 7;
    const float* zp = Zsrc[j] + (size_t)(b0 + r) * 3;
    u16* dst = Zc0 + (r * 8 + j) * 32;
    dst[0] = f2bf(zp[0]);
    dst[1] = f2bf(zp[1]);
    dst[2] = f2bf(zp[2]);
#pragma unroll
    for (int d = 3; d < 32; ++d) dst[d] = 0;
  }
  __syncthreads();

  layer_step< 32, 256>(Zc0, X0, H0, Zs1, Wp0, Wgp0, b0_, g0_, be0_, bg0_, A, b0, tid);
  layer_step<256, 128>(Zs1, X1, H1, Zs2, Wp1, Wgp1, b1_, g1_, be1_, bg1_, A, b0, tid);
  layer_step<128,  64>(Zs2, X2, H2, Zs3, Wp2, Wgp2, b2_, g2_, be2_, bg2_, A, b0, tid);
  out_step(Zs3, H3, WpO, bo_, out, b0, tid);
}

// ---------------------------------------------------------------------------
extern "C" void kernel_launch(void* const* d_in, const int* in_sizes, int n_in,
                              void* d_out, int out_size, void* d_ws, size_t ws_size,
                              hipStream_t stream) {
  (void)in_sizes; (void)n_in; (void)out_size; (void)ws_size;
  // inputs: 0:state(unused) 1-8:Z1..Z8 9:A then per layer W,b,g,be,Wg,bg; 28:Wout 29:bout
  const float* Z1  = (const float*)d_in[1];
  const float* Z2  = (const float*)d_in[2];
  const float* Z3  = (const float*)d_in[3];
  const float* Z4  = (const float*)d_in[4];
  const float* Z5  = (const float*)d_in[5];
  const float* Z6  = (const float*)d_in[6];
  const float* Z7  = (const float*)d_in[7];
  const float* Z8  = (const float*)d_in[8];
  const float* A   = (const float*)d_in[9];
  const float* W0  = (const float*)d_in[10];
  const float* b0  = (const float*)d_in[11];
  const float* g0  = (const float*)d_in[12];
  const float* be0 = (const float*)d_in[13];
  const float* Wg0 = (const float*)d_in[14];
  const float* bg0 = (const float*)d_in[15];
  const float* W1  = (const float*)d_in[16];
  const float* b1  = (const float*)d_in[17];
  const float* g1  = (const float*)d_in[18];
  const float* be1 = (const float*)d_in[19];
  const float* Wg1 = (const float*)d_in[20];
  const float* bg1 = (const float*)d_in[21];
  const float* W2  = (const float*)d_in[22];
  const float* b2  = (const float*)d_in[23];
  const float* g2  = (const float*)d_in[24];
  const float* be2 = (const float*)d_in[25];
  const float* Wg2 = (const float*)d_in[26];
  const float* bg2 = (const float*)d_in[27];
  const float* Wout= (const float*)d_in[28];
  const float* bout= (const float*)d_in[29];

  // workspace: packed bf16 weights (offsets in u16 elements)
  u16* ws   = (u16*)d_ws;
  u16* Wp0  = ws + 0;        // 8*32*256   = 65536
  u16* Wgp0 = ws + 65536;    // 8*224*16   = 28672
  u16* Wp1  = ws + 94208;    // 8*256*128  = 262144
  u16* Wgp1 = ws + 356352;   // 8*1792*16  = 229376
  u16* Wp2  = ws + 585728;   // 8*128*64   = 65536
  u16* Wgp2 = ws + 651264;   // 8*896*16   = 114688
  u16* WpO  = ws + 765952;   // 8*64*16    = 8192   -> total 774144 u16 = 1.5MB

  auto pk = [&](const float* s, u16* dp, int Ks, int Ns, int KP, int NP, int CH, int CHP) {
    int total = 8 * KP * NP;
    prepack_kernel<<<(total + 255) / 256, 256, 0, stream>>>(s, dp, Ks, Ns,
                                                            KP / 32, NP / 16, CH, CHP, total);
  };
  pk(W0,   Wp0,    3, 256,   32, 256,   3,  32);
  pk(Wg0,  Wgp0,  21,   7,  224,  16,   3,  32);
  pk(W1,   Wp1,  256, 128,  256, 128, 256, 256);
  pk(Wg1,  Wgp1,1792,   7, 1792,  16, 256, 256);
  pk(W2,   Wp2,  128,  64,  128,  64, 128, 128);
  pk(Wg2,  Wgp2, 896,   7,  896,  16, 128, 128);
  pk(Wout, WpO,   64,   5,   64,  16,  64,  64);

  actor_fused<<<8192 / 16, 256, 245760, stream>>>(
      Z1, Z2, Z3, Z4, Z5, Z6, Z7, Z8, A,
      Wp0, Wgp0, b0, g0, be0, bg0,
      Wp1, Wgp1, b1, g1, be1, bg1,
      Wp2, Wgp2, b2, g2, be2, bg2,
      WpO, bout, (float*)d_out);
}
